// Recurrence_73804718014820
// MI455X (gfx1250) — matrix operand order repaired and checked by reading, not compiled, not run on hardware
//
#include <hip/hip_runtime.h>
#include <stdint.h>

// ---------------- problem constants ----------------
#define T_STEPS 32
#define NBATCH  256
#define RAWC    4155            // per-(t,n) raw_inputs stride
#define OUTC    43
#define LINES_OFF 4096
#define MASK_OFF  4136
#define RES_OFF   4146
#define ACT_OFF   4150

typedef __attribute__((ext_vector_type(16))) __bf16 v16bf;
typedef __attribute__((ext_vector_type(8)))  float  v8f;

// ---------------- bf16 helpers (bit-exact, no reliance on native __bf16 math) --
static __device__ __forceinline__ float bf2f(__bf16 b) {
  unsigned int u = ((unsigned int)__builtin_bit_cast(unsigned short, b)) << 16;
  return __builtin_bit_cast(float, u);
}
static __device__ __forceinline__ __bf16 f2bf(float f) {
  unsigned int u = __builtin_bit_cast(unsigned int, f);
  unsigned int r = (u + 0x7FFFu + ((u >> 16) & 1u)) >> 16;
  return __builtin_bit_cast(__bf16, (unsigned short)r);
}
static __device__ __forceinline__ float sigmoidf_(float x) { return 1.f / (1.f + __expf(-x)); }

static __device__ __forceinline__ v8f vzero8() {
  v8f v;
#pragma unroll
  for (int i = 0; i < 8; ++i) v[i] = 0.f;
  return v;
}

// ---------------- WMMA fragment loaders (A/B use contiguous k = 16h + s, wave32) ----------
// A (16x32 bf16, M x K): lane = M (0..15) + 16*(K-half); VGPR j holds packed K pair.
static __device__ __forceinline__ v16bf load_A16(const __bf16* X, int ld, int k0, int lane) {
  int m = lane & 15, kh = (lane >> 4) & 1;
  const __bf16* row = X + (size_t)m * ld + k0 + 16 * kh;
  v16bf a;
#pragma unroll
  for (int j = 0; j < 8; ++j) {
    int k = 2 * j;
    a[2 * j]     = row[k];
    a[2 * j + 1] = row[k + 1];
  }
  return a;
}
// B (32x16 bf16, K x N) where the weight is stored row-major [Ncols][K]
// (i.e. "transposed" so one lane's 16 K-values are contiguous).
static __device__ __forceinline__ v16bf load_Bt(const __bf16* W, int ld, int n0, int k0, int lane) {
  int n = lane & 15, kh = (lane >> 4) & 1;
  const __bf16* col = W + (size_t)(n0 + n) * ld + k0 + 16 * kh;
  v16bf b;
#pragma unroll
  for (int j = 0; j < 16; ++j) b[j] = col[j];
  return b;
}
// D/C (16x16 f32): lane = N + 16*(M>=8); VGPR j = M%8.
static __device__ __forceinline__ void store_D(float* Y, int ld, int n0, v8f d, int lane) {
  int n = lane & 15, mh = (lane >> 4) & 1;
#pragma unroll
  for (int j = 0; j < 8; ++j) Y[(size_t)(8 * mh + j) * ld + n0 + n] = d[j];
}
static __device__ __forceinline__ v8f wmma_bf(v16bf a, v16bf b, v8f c) {
  return __builtin_amdgcn_wmma_f32_16x16x32_bf16(false, a, false, b, (short)0, c, false, false);
}

// ---------------- kernel argument bundle ----------------
struct Args {
  const float* raw;   const float* hxs;
  const float* gbihf; const float* gbhhf; const float* gbihb; const float* gbhhb;
  const float* betaW; const float* betab;
  const float* kb;    const float* conv_bias;
  const float* invW;  const float* invb;  const float* z1b;
  const float* upW;   const float* upb;   const float* dgW;   const float* dgb;
  const float* embL;  const float* actb;  const float* crW;   const float* crb;
  const float* masks; const int* thr;
  const __bf16* Wihf; const __bf16* Whhf; const __bf16* Wihb; const __bf16* Whhb;
  const __bf16* kWT;  const __bf16* z1WT; const __bf16* actWT; const __bf16* Mbf;
  __bf16* kernws;
  float* out;
};

// ---------------- LDS phase union (max 49152 B) ----------------
struct GruP  { __bf16 x[16][128]; __bf16 hbf[16][128]; float h[16][128];
               float grz[16][256]; float ginn[16][128]; float ghn[16][128]; };
struct ConvP { __bf16 kern[9216]; __bf16 obs[16][18][18]; };
struct ZP    { __bf16 zbf[16][320]; __bf16 zact[16][576]; float alog[16][112]; };
union PhaseU { GruP g; ConvP c; ZP z; };

// ---------------- preprocessing kernels ----------------
__global__ void k_cvt_bf16(const float* s, __bf16* d, int n) {
  int i = blockIdx.x * blockDim.x + threadIdx.x;
  if (i < n) d[i] = f2bf(s[i]);
}
__global__ void k_transpose_bf16(const float* s, __bf16* d, int K, int C) { // s[K][C] -> d[C][K]
  int i = blockIdx.x * blockDim.x + threadIdx.x;
  if (i < K * C) { int k = i / C, c = i % C; d[(size_t)c * K + k] = f2bf(s[i]); }
}
__global__ void k_actWT(const float* s, __bf16* d) { // actW (576,100) -> d[112][576], zero-padded
  int i = blockIdx.x * blockDim.x + threadIdx.x;
  if (i < 112 * 576) { int c = i / 576, k = i % 576; d[i] = (c < 100) ? f2bf(s[k * 100 + c]) : f2bf(0.f); }
}
__global__ void k_buildM(const float* raw, const float* emb, __bf16* M) {
  int i = blockIdx.x * blockDim.x + threadIdx.x;
  if (i < 256 * 10 * 128) {
    int n = i / 1280, l = (i / 128) % 10, k = i & 127;
    float s = 0.f;
    for (int v = 0; v < 4; ++v) {
      int tok = (int)raw[(size_t)n * RAWC + LINES_OFF + l * 4 + v];
      tok = tok < 0 ? 0 : (tok > 19 ? 19 : tok);
      s += emb[((size_t)v * 20 + tok) * 128 + k];
    }
    M[i] = f2bf(s);
  }
}

// ---------------- fused T=32 recurrence: 1 WG = 16 samples, 4 wave32 ----------
__global__ __launch_bounds__(128) void fused_recurrence(Args ar) {
  __shared__ PhaseU U;
  __shared__ float h1acc[16][64];
  __shared__ float Bv[16][10][2][4];
  __shared__ float Pm[16][20][4];
  __shared__ int pc[16], lc[16], lnewc[16];
  __shared__ int Aprev[16][3];

  const int tid  = threadIdx.x;
  const int lane = tid & 31;
  const int wave = tid >> 5;
  const int s0   = blockIdx.x * 16;

  if (tid < 16) {
    int n = tid;
    int p = (int)ar.hxs[(size_t)(s0 + n) * 560 + 554];
    pc[n] = p < 0 ? 0 : (p > 9 ? 9 : p);
    int l = (int)ar.hxs[(size_t)(s0 + n) * 560 + 556];
    lc[n] = l < 0 ? 0 : (l > 2 ? 2 : l);
    const float* alast = ar.raw + ((size_t)(T_STEPS - 1) * NBATCH + s0 + n) * RAWC + ACT_OFF;
    for (int j = 0; j < 3; ++j) Aprev[n][j] = (int)alast[j];
  }
  __syncthreads();

  for (int t = 0; t < T_STEPS; ++t) {
    // ============ bidirectional GRU over pointer-rolled task encoding ============
    for (int dir = 0; dir < 2; ++dir) {
      const __bf16* Wih = dir ? ar.Wihb : ar.Wihf;
      const __bf16* Whh = dir ? ar.Whhb : ar.Whhf;
      const float*  bih = dir ? ar.gbihb : ar.gbihf;
      const float*  bhh = dir ? ar.gbhhb : ar.gbhhf;
      for (int idx = tid; idx < 2048; idx += 128) {
        int n = idx >> 7, k = idx & 127;
        U.g.h[n][k]   = 0.f;
        U.g.hbf[n][k] = f2bf(0.f);
        int line = dir ? (pc[n] + 9) % 10 : pc[n];
        U.g.x[n][k] = ar.Mbf[((size_t)(s0 + n) * 10 + line) * 128 + k];
      }
      __syncthreads();
      for (int st = 0; st < 10; ++st) {
        const int linecur = dir ? (9 - st) : st;
        // 32 WMMA tile-jobs: 16 r/z tiles (gi+gh chained in C), 8 gi_n, 8 gh_n
        for (int job = wave; job < 32; job += 4) {
          v8f acc = vzero8();
          if (job < 16) {
            int n0 = job * 16;
#pragma unroll
            for (int ks = 0; ks < 4; ++ks)
              acc = wmma_bf(load_A16(&U.g.x[0][0], 128, ks * 32, lane),
                            load_Bt(Wih, 128, n0, ks * 32, lane), acc);
#pragma unroll
            for (int ks = 0; ks < 4; ++ks)
              acc = wmma_bf(load_A16(&U.g.hbf[0][0], 128, ks * 32, lane),
                            load_Bt(Whh, 128, n0, ks * 32, lane), acc);
            store_D(&U.g.grz[0][0], 256, n0, acc, lane);
          } else if (job < 24) {
            int n0 = (job - 16) * 16;
#pragma unroll
            for (int ks = 0; ks < 4; ++ks)
              acc = wmma_bf(load_A16(&U.g.x[0][0], 128, ks * 32, lane),
                            load_Bt(Wih, 128, 256 + n0, ks * 32, lane), acc);
            store_D(&U.g.ginn[0][0], 128, n0, acc, lane);
          } else {
            int n0 = (job - 24) * 16;
#pragma unroll
            for (int ks = 0; ks < 4; ++ks)
              acc = wmma_bf(load_A16(&U.g.hbf[0][0], 128, ks * 32, lane),
                            load_Bt(Whh, 128, 256 + n0, ks * 32, lane), acc);
            store_D(&U.g.ghn[0][0], 128, n0, acc, lane);
          }
        }
        __syncthreads();
        // gate nonlinearity + hidden update + gather next position's input
        for (int idx = tid; idx < 2048; idx += 128) {
          int n = idx >> 7, k = idx & 127;
          float r  = sigmoidf_(U.g.grz[n][k]       + bih[k]       + bhh[k]);
          float zg = sigmoidf_(U.g.grz[n][128 + k] + bih[128 + k] + bhh[128 + k]);
          float ng = tanhf(U.g.ginn[n][k] + bih[256 + k] + r * (U.g.ghn[n][k] + bhh[256 + k]));
          float hn = (1.f - zg) * ng + zg * U.g.h[n][k];
          U.g.h[n][k]   = hn;
          U.g.hbf[n][k] = f2bf(hn);
          if (st < 9) {
            int nl = dir ? (pc[n] + 8 - st) % 10 : (pc[n] + st + 1) % 10;
            U.g.x[n][k] = ar.Mbf[((size_t)(s0 + n) * 10 + nl) * 128 + k];
          }
        }
        __syncthreads();
        // beta head: B = sigmoid(h @ betaW + b) * maskP[p]
        if (tid < 64) {
          int n = tid >> 2, e = tid & 3;
          float s = ar.betab[e];
          for (int k = 0; k < 128; ++k) s += U.g.h[n][k] * ar.betaW[k * 4 + e];
          float sg = sigmoidf_(s);
          int j20 = (dir * 10 + linecur + pc[n]) % 20;
          float mv;
          if (linecur == 0 && dir == 0) mv = 0.f;
          else mv = (j20 < 10) ? ar.raw[(size_t)(s0 + n) * RAWC + MASK_OFF + j20] : 0.f;
          Bv[n][linecur][dir][e] = sg * mv;
        }
        __syncthreads();
      }
    }

    // ============ stick-breaking pointer distribution P (per sample) ============
    if (tid < 16) {
      int n = tid;
      float B2[20][4], Cc[20][4];
      for (int k = 0; k < 10; ++k)
        for (int e = 0; e < 4; ++e) {
          B2[2 * k][e]     = Bv[n][k][0][e];
          B2[2 * k + 1][e] = Bv[n][9 - k][1][e];
        }
      float run[4] = {1.f, 1.f, 1.f, 1.f};
      for (int j = 0; j < 20; ++j) {
        int jm = (j + 19) % 20;
        for (int e = 0; e < 4; ++e) { run[e] *= (1.f - B2[jm][e]); Cc[j][e] = run[e]; }
      }
      for (int j = 0; j < 20; ++j) {
        float fi = (j == 0) ? 0.1f : 0.f;
        for (int e = 0; e < 4; ++e) Cc[j][e] = ((1.f - fi) * B2[j][e] + fi) * Cc[j][e];
      }
      for (int k = 0; k < 10; ++k)
        for (int e = 0; e < 4; ++e) Pm[n][k][e] = Cc[2 * (9 - k) + 1][e];
      for (int k = 10; k < 20; ++k)
        for (int e = 0; e < 4; ++e) Pm[n][k][e] = Cc[2 * (k - 10)][e];
    }
    for (int idx = tid; idx < 1024; idx += 128) (&h1acc[0][0])[idx] = 0.f;
    // gather m = M[n, p[n]] for kern hyper-GEMM
    for (int idx = tid; idx < 2048; idx += 128) {
      int n = idx >> 7, k = idx & 127;
      U.g.x[n][k] = ar.Mbf[((size_t)(s0 + n) * 10 + pc[n]) * 128 + k];
    }
    __syncthreads();

    // ============ dynamic-conv kernel hyper-GEMM: (16x128)@(128x9216) ===========
    v16bf mfrag[4];
#pragma unroll
    for (int ks = 0; ks < 4; ++ks) mfrag[ks] = load_A16(&U.g.x[0][0], 128, ks * 32, lane);
    __bf16* kernws = ar.kernws + (size_t)blockIdx.x * 16 * 9216;
    for (int jt = wave; jt < 576; jt += 4) {
      v8f acc = vzero8();
#pragma unroll
      for (int ks = 0; ks < 4; ++ks)
        acc = wmma_bf(mfrag[ks], load_Bt(ar.kWT, 128, jt * 16, ks * 32, lane), acc);
      int nn = lane & 15, mh = (lane >> 4) & 1;
      float kbv = ar.kb[jt * 16 + nn];
#pragma unroll
      for (int j = 0; j < 8; ++j)
        kernws[(size_t)(8 * mh + j) * 9216 + jt * 16 + nn] = f2bf(acc[j] + kbv);
    }
    __threadfence();
    __syncthreads();

    // ============ per-sample dynamic conv + relu + spatial sum ============
    for (int s = 0; s < 16; ++s) {
      const uint32_t* ksrc = (const uint32_t*)(kernws + (size_t)s * 9216);
      uint32_t* kdst = (uint32_t*)&U.c.kern[0];
      for (int i = tid; i < 4608; i += 128) kdst[i] = ksrc[i];
      const float* ob = ar.raw + ((size_t)t * NBATCH + s0 + s) * RAWC;
      for (int i = tid; i < 5184; i += 128) {
        int c = i / 324, r = (i / 18) % 18, q = i % 18;
        float v = 0.f;
        if (r >= 1 && r <= 16 && q >= 1 && q <= 16) v = ob[c * 256 + (r - 1) * 16 + (q - 1)];
        U.c.obs[c][r][q] = f2bf(v);
      }
      if (s < 15 && tid == 0)
        __builtin_prefetch(ar.raw + ((size_t)t * NBATCH + s0 + s + 1) * RAWC, 0, 1);
      __syncthreads();
      for (int ch = 0; ch < 64; ++ch) {
        float bia = ar.conv_bias[ch];
        float part = 0.f;
#pragma unroll
        for (int rep = 0; rep < 2; ++rep) {
          int xy = tid + rep * 128;
          int x = xy >> 4, y = xy & 15;
          float acc = bia;
          for (int c = 0; c < 16; ++c)
#pragma unroll
            for (int kx = 0; kx < 3; ++kx)
#pragma unroll
              for (int ky = 0; ky < 3; ++ky)
                acc += bf2f(U.c.obs[c][x + kx][y + ky]) *
                       bf2f(U.c.kern[ch * 144 + c * 9 + kx * 3 + ky]);
          part += fmaxf(acc, 0.f);
        }
        atomicAdd(&h1acc[s][ch], part);
      }
      __syncthreads();
    }

    // ============ z = [m, h1, inv] (bf16) ============
    for (int idx = tid; idx < 16 * 320; idx += 128) {
      int n = idx / 320, k = idx % 320;
      float v;
      if (k < 128) v = bf2f(ar.Mbf[((size_t)(s0 + n) * 10 + pc[n]) * 128 + k]);
      else if (k < 192) v = h1acc[n][k - 128];
      else {
        int kk = k - 192;
        const float* res = ar.raw + ((size_t)t * NBATCH + s0 + n) * RAWC + RES_OFF;
        float s2 = ar.invb[kk];
#pragma unroll
        for (int i = 0; i < 4; ++i) s2 += res[i] * ar.invW[i * 128 + kk];
        v = fmaxf(s2, 0.f);
      }
      U.z.zbf[n][k] = f2bf(v);
    }
    __syncthreads();

    // ============ z1 = relu(z @ z1W + b): (16x320)@(320x512) WMMA ============
    for (int jt = wave; jt < 32; jt += 4) {
      v8f acc = vzero8();
#pragma unroll
      for (int ks = 0; ks < 10; ++ks)
        acc = wmma_bf(load_A16(&U.z.zbf[0][0], 320, ks * 32, lane),
                      load_Bt(ar.z1WT, 320, jt * 16, ks * 32, lane), acc);
      int nn = lane & 15, mh = (lane >> 4) & 1;
      float b = ar.z1b[jt * 16 + nn];
#pragma unroll
      for (int j = 0; j < 8; ++j)
        U.z.zact[8 * mh + j][jt * 16 + nn] = f2bf(fmaxf(acc[j] + b, 0.f));
    }
    // append emb_low (uses OLD l and OLD A_prev)
    for (int idx = tid; idx < 1024; idx += 128) {
      int n = idx >> 6, k = idx & 63;
      int lo = lc[n];
      int lower = Aprev[n][lo];
      lower = lower < 0 ? 0 : (lower > 7 ? 7 : lower);
      U.z.zact[n][512 + k] = f2bf(ar.embL[((size_t)lo * 8 + lower) * 64 + k]);
    }
    __syncthreads();

    // ============ actor head: (16x576)@(576x112pad) WMMA ============
    for (int jt = wave; jt < 7; jt += 4) {
      v8f acc = vzero8();
#pragma unroll
      for (int ks = 0; ks < 18; ++ks)
        acc = wmma_bf(load_A16(&U.z.zact[0][0], 576, ks * 32, lane),
                      load_Bt(ar.actWT, 576, jt * 16, ks * 32, lane), acc);
      store_D(&U.z.alog[0][0], 112, jt * 16, acc, lane);
    }
    if (tid < 16) { // l_new from OLD A_prev
      int n = tid, cnt = 0; bool aboveall = true;
      for (int j = 0; j < 3; ++j)
        if (Aprev[n][j] >= 0) { cnt++; if (!(Aprev[n][j] > ar.thr[j])) aboveall = false; }
      lnewc[n] = aboveall ? (cnt % 3) : 0;
    }
    __syncthreads();

    // ============ per-sample epilogue: softmaxes, gating, value, carry ============
    if (tid < 16) {
      int n = tid;
      float* o = ar.out + ((size_t)t * NBATCH + s0 + n) * OUTC;
      int row = lnewc[n];
      float lg[20], mx = -1e30f;
      for (int j = 0; j < 20; ++j) {
        lg[j] = U.z.alog[n][row * 20 + j] + ar.actb[row * 20 + j];
        mx = fmaxf(mx, lg[j]);
      }
      float se = 0.f;
      for (int j = 0; j < 20; ++j) { lg[j] = __expf(lg[j] - mx); se += lg[j]; }
      float sm = 0.f;
      for (int j = 0; j < 20; ++j) { lg[j] = lg[j] / se * ar.masks[row * 20 + j]; sm += lg[j]; }
      float ism = 1.f / sm;
      for (int j = 0; j < 20; ++j) o[j] = lg[j] * ism;

      float ul[4] = {ar.upb[0], ar.upb[1], ar.upb[2], ar.upb[3]};
      float dl[2] = {ar.dgb[0], ar.dgb[1]};
      for (int k = 0; k < 320; ++k) {
        float zv = bf2f(U.z.zbf[n][k]);
#pragma unroll
        for (int e = 0; e < 4; ++e) ul[e] += zv * ar.upW[k * 4 + e];
        dl[0] += zv * ar.dgW[k * 2 + 0];
        dl[1] += zv * ar.dgW[k * 2 + 1];
      }
      float um = fmaxf(fmaxf(ul[0], ul[1]), fmaxf(ul[2], ul[3]));
      float u[4], us = 0.f;
      for (int e = 0; e < 4; ++e) { u[e] = __expf(ul[e] - um); us += u[e]; }
      for (int e = 0; e < 4; ++e) u[e] /= us;
      float dm = fmaxf(dl[0], dl[1]);
      float e0 = __expf(dl[0] - dm), e1 = __expf(dl[1] - dm);

      const float* act = ar.raw + ((size_t)t * NBATCH + s0 + n) * RAWC + ACT_OFF;
      float dg = act[4];
      float dp[20], ds = 0.f;
      for (int k = 0; k < 20; ++k) {
        float s3 = 0.f;
#pragma unroll
        for (int e = 0; e < 4; ++e) s3 += Pm[n][k][e] * u[e];
        s3 = dg * s3 + (1.f - dg) * ((k == 10) ? 1.f : 0.f);
        dp[k] = s3; ds += s3;
      }
      float ids = 1.f / ds;
      for (int k = 0; k < 20; ++k) o[20 + k] = dp[k] * ids;
      o[40] = e0 / (e0 + e1);
      o[41] = e1 / (e0 + e1);
      float vv = ar.crb[0];
      for (int k = 0; k < 512; ++k) vv += bf2f(U.z.zact[n][k]) * ar.crW[k];
      o[42] = vv;

      // carry update (A_eff, l, p)
      for (int j = 0; j < 3; ++j) if (j >= row) Aprev[n][j] = (int)act[j];
      lc[n] = row;
      int pn = pc[n] + (int)act[3] - 10;
      pc[n] = pn < 0 ? 0 : (pn > 9 ? 9 : pn);
    }
    __syncthreads();
  }
}

// ---------------- host-side launcher ----------------
extern "C" void kernel_launch(void* const* d_in, const int* in_sizes, int n_in,
                              void* d_out, int out_size, void* d_ws, size_t ws_size,
                              hipStream_t stream) {
  (void)in_sizes; (void)n_in; (void)out_size; (void)ws_size;
  const float* raw      = (const float*)d_in[0];
  const float* hxs      = (const float*)d_in[1];
  const float* emb_task = (const float*)d_in[2];
  const float* gWihf    = (const float*)d_in[3];
  const float* gWhhf    = (const float*)d_in[4];
  const float* gbihf    = (const float*)d_in[5];
  const float* gbhhf    = (const float*)d_in[6];
  const float* gWihb    = (const float*)d_in[7];
  const float* gWhhb    = (const float*)d_in[8];
  const float* gbihb    = (const float*)d_in[9];
  const float* gbhhb    = (const float*)d_in[10];
  const float* betaW    = (const float*)d_in[11];
  const float* betab    = (const float*)d_in[12];
  const float* kW       = (const float*)d_in[13];
  const float* kb       = (const float*)d_in[14];
  const float* conv_bias= (const float*)d_in[15];
  const float* invW     = (const float*)d_in[16];
  const float* invb     = (const float*)d_in[17];
  const float* z1W      = (const float*)d_in[18];
  const float* z1b      = (const float*)d_in[19];
  const float* upW      = (const float*)d_in[20];
  const float* upb      = (const float*)d_in[21];
  const float* dgW      = (const float*)d_in[22];
  const float* dgb      = (const float*)d_in[23];
  const float* embL     = (const float*)d_in[24];
  const float* actW     = (const float*)d_in[25];
  const float* actb     = (const float*)d_in[26];
  const float* crW      = (const float*)d_in[27];
  const float* crb      = (const float*)d_in[28];
  const float* masks    = (const float*)d_in[29];
  const int*   thr      = (const int*)d_in[30];

  char* ws = (char*)d_ws;
  __bf16* Wihf_bf  = (__bf16*)(ws + 0);
  __bf16* Whhf_bf  = (__bf16*)(ws + 98304);
  __bf16* Wihb_bf  = (__bf16*)(ws + 196608);
  __bf16* Whhb_bf  = (__bf16*)(ws + 294912);
  __bf16* kWT_bf   = (__bf16*)(ws + 393216);
  __bf16* z1WT_bf  = (__bf16*)(ws + 2752512);
  __bf16* actWT_bf = (__bf16*)(ws + 3080192);
  __bf16* Mbf      = (__bf16*)(ws + 3209216);
  __bf16* kernws   = (__bf16*)(ws + 3864576);

  // per-launch weight conversion / transposition into bf16 WMMA-friendly layouts
  k_cvt_bf16<<<192, 256, 0, stream>>>(gWihf, Wihf_bf, 49152);
  k_cvt_bf16<<<192, 256, 0, stream>>>(gWhhf, Whhf_bf, 49152);
  k_cvt_bf16<<<192, 256, 0, stream>>>(gWihb, Wihb_bf, 49152);
  k_cvt_bf16<<<192, 256, 0, stream>>>(gWhhb, Whhb_bf, 49152);
  k_transpose_bf16<<<4608, 256, 0, stream>>>(kW, kWT_bf, 128, 9216);
  k_transpose_bf16<<<640,  256, 0, stream>>>(z1W, z1WT_bf, 320, 512);
  k_actWT<<<252, 256, 0, stream>>>(actW, actWT_bf);
  k_buildM<<<1280, 256, 0, stream>>>(raw, emb_task, Mbf);

  Args a;
  a.raw = raw; a.hxs = hxs;
  a.gbihf = gbihf; a.gbhhf = gbhhf; a.gbihb = gbihb; a.gbhhb = gbhhb;
  a.betaW = betaW; a.betab = betab;
  a.kb = kb; a.conv_bias = conv_bias;
  a.invW = invW; a.invb = invb; a.z1b = z1b;
  a.upW = upW; a.upb = upb; a.dgW = dgW; a.dgb = dgb;
  a.embL = embL; a.actb = actb; a.crW = crW; a.crb = crb;
  a.masks = masks; a.thr = thr;
  a.Wihf = Wihf_bf; a.Whhf = Whhf_bf; a.Wihb = Wihb_bf; a.Whhb = Whhb_bf;
  a.kWT = kWT_bf; a.z1WT = z1WT_bf; a.actWT = actWT_bf; a.Mbf = Mbf;
  a.kernws = kernws;
  a.out = (float*)d_out;

  fused_recurrence<<<16, 128, 0, stream>>>(a);
}
